// UNet_GNN_pass_29454885715979
// MI455X (gfx1250) — matrix operand order repaired and checked by reading, not compiled, not run on hardware
//
#include <hip/hip_runtime.h>

// ---------------------------------------------------------------------------
// CDNA5 (gfx1250) wave32 WMMA UNet+GNN forward pass.
// All conv / convT / SAGE-linear layers run through one implicit-GEMM kernel
// using v_wmma_f32_16x16x32_bf16 (f32 accumulate), with:
//   - async-to-LDS staging of weight tiles (ASYNCcnt path)
//   - Tensor Data Mover staging of dense-GEMM B tiles (TENSORcnt path)
//   - vectorized ds fragment loads, global_prefetch on weight stream.
// Activations live in [C,B,H,W] layout so skip-concats are free.
// ---------------------------------------------------------------------------

typedef __attribute__((ext_vector_type(16))) __bf16   v16bf;
typedef __attribute__((ext_vector_type(8)))  __bf16   v8bf;
typedef __attribute__((ext_vector_type(8)))  float    v8f;
typedef __attribute__((ext_vector_type(4)))  unsigned v4u;
typedef __attribute__((ext_vector_type(8)))  unsigned v8u;

#define TILE_M 32
#define TILE_N 128
#define TILE_K 32

struct GemmConvParams {
    const float*  src;   // activations [Ci, B, H, W] fp32
    const __bf16* wA;    // weights [Co, K], K = Ci*KH*KW, bf16
    float*        dst;   // output [Co, Ntot], Ntot = B*Ho*Wo  (== [Co,B,Ho,Wo])
    int Co, Ci, B, H, W, Ho, Wo;
    int KH, KW;          // 3 (conv) or 1 (dense GEMM)
    int stride;          // output-coord stride (conv); 1 for convT
    int dil;             // lhs dilation (convT); 1 for conv
    int pad;             // low padding
};

__launch_bounds__(256)
__global__ void conv_wmma_kernel(GemmConvParams p) {
    // A tile: [m][k] row-major (64B rows -> b128 async segments)
    __shared__ __bf16 lA[TILE_M * TILE_K];
    // B tile in fragment-major order: [(n*2 + k_hi)*16 + k_lo]
    __shared__ __bf16 lBp[TILE_N * TILE_K];
    // TDM staging for dense-GEMM B tiles (fp32, [k][n])
    __shared__ float  lSrc[TILE_K * TILE_N];

    const int Ntot = p.B * p.Ho * p.Wo;
    const int Ktot = p.Ci * p.KH * p.KW;
    const int HoWo = p.Ho * p.Wo;
    const int KHW  = p.KH * p.KW;

    const int tid  = threadIdx.x;
    const int wave = tid >> 5;      // 0..7
    const int lane = tid & 31;
    const int wm   = wave >> 2;     // 0..1 : 16-row subtile
    const int wn   = wave & 3;      // 0..3 : 32-col subtile (2 x 16)
    const int m0   = blockIdx.y * TILE_M;
    const int n0   = blockIdx.x * TILE_N;

    v8f acc0 = {}, acc1 = {};

    for (int k0 = 0; k0 < Ktot; k0 += TILE_K) {
        const bool full = (m0 + TILE_M <= p.Co) && (k0 + TILE_K <= Ktot);

        // ---- stage A (weights) tile: TILE_M x TILE_K -----------------------
        if (full) {
            // async DMA: 32 rows x 4 x 16B segments, tracked by ASYNCcnt
            if (tid < 128) {
                int r = tid >> 2, seg = tid & 3;
                const __bf16* g =
                    p.wA + (size_t)(m0 + r) * Ktot + k0 + seg * 8;
                unsigned lds = (unsigned)(size_t)&lA[r * TILE_K + seg * 8];
                asm volatile("global_load_async_to_lds_b128 %0, %1, off"
                             :: "v"(lds), "v"((unsigned long long)(size_t)g)
                             : "memory");
            }
        } else {
            for (int i = tid; i < TILE_M * TILE_K; i += 256) {
                int m = i / TILE_K, k = i % TILE_K;
                int gm = m0 + m, gk = k0 + k;
                lA[i] = (gm < p.Co && gk < Ktot)
                            ? p.wA[(size_t)gm * Ktot + gk] : (__bf16)0.0f;
            }
        }
        // prefetch next weight K-slab (global_prefetch_b8)
        if (k0 + TILE_K < Ktot && tid < TILE_M) {
            int gm = m0 + tid;
            if (gm < p.Co)
                __builtin_prefetch(&p.wA[(size_t)gm * Ktot + k0 + TILE_K], 0, 1);
        }

        // ---- stage B tile: TILE_K x TILE_N ---------------------------------
        const bool use_tdm = (KHW == 1) && full && (n0 + TILE_N <= Ntot);
        if (use_tdm) {
            // Dense GEMM: B tile is a rectangular 2D region of src ->
            // single Tensor Data Mover transfer (TENSORcnt path).
            if (wave == 0) {
                unsigned long long ga =
                    (unsigned long long)(size_t)(p.src + (size_t)k0 * Ntot + n0);
                v4u g0;
                g0[0] = 1u;                                     // count=1 (valid)
                g0[1] = (unsigned)(size_t)&lSrc[0];             // lds_addr
                g0[2] = (unsigned)ga;                           // global_addr lo
                g0[3] = (unsigned)((ga >> 32) & 0x1FFFFFFu)     // global_addr hi
                        | (2u << 30);                           // type=2 (image)
                unsigned td0 = (unsigned)Ntot;                  // tensor dim0
                unsigned td1 = (unsigned)Ktot;                  // tensor dim1
                v8u g1;
                g1[0] = (2u << 16);                             // data_size=4B
                g1[1] = (td0 & 0xFFFFu) << 16;                  // td0[15:0]@48
                g1[2] = (td0 >> 16) | ((td1 & 0xFFFFu) << 16);  // td0 hi | td1 lo
                g1[3] = (td1 >> 16) | ((unsigned)TILE_N << 16); // td1 hi | tile0
                g1[4] = (unsigned)TILE_K;                       // tile_dim1=32
                g1[5] = td0;                                    // dim0_stride lo
                g1[6] = 0u;                                     // stride hi
                g1[7] = 0u;
                asm volatile("tensor_load_to_lds %0, %1"
                             :: "s"(g0), "s"(g1) : "memory");
                asm volatile("s_wait_tensorcnt 0" ::: "memory");
            }
            asm volatile("s_wait_asynccnt 0" ::: "memory");     // A tile done
            __syncthreads();
            // fp32 staged tile -> fragment-major bf16
            for (int i = tid; i < TILE_K * TILE_N; i += 256) {
                int k = i / TILE_N, n = i % TILE_N;
                lBp[(n * 2 + (k >> 4)) * 16 + (k & 15)] =
                    (__bf16)lSrc[k * TILE_N + n];
            }
            __syncthreads();
        } else {
            // implicit im2col gather (handles padding / dilation / edges)
            for (int i = tid; i < TILE_K * TILE_N; i += 256) {
                int k = i / TILE_N, n = i % TILE_N;
                int gk = k0 + k, gn = n0 + n;
                float v = 0.0f;
                if (gk < Ktot && gn < Ntot) {
                    int ci = gk / KHW, rem = gk % KHW;
                    int kh = rem / p.KW, kw = rem % p.KW;
                    int b  = gn / HoWo, r2 = gn % HoWo;
                    int oh = r2 / p.Wo, ow = r2 % p.Wo;
                    int ihv = oh * p.stride - p.pad + kh;
                    int iwv = ow * p.stride - p.pad + kw;
                    if (p.dil == 1) {
                        if (ihv >= 0 && ihv < p.H && iwv >= 0 && iwv < p.W)
                            v = p.src[(size_t)((ci * p.B + b) * p.H + ihv) * p.W + iwv];
                    } else {
                        if (ihv >= 0 && iwv >= 0 &&
                            (ihv % p.dil) == 0 && (iwv % p.dil) == 0) {
                            int ih = ihv / p.dil, iw = iwv / p.dil;
                            if (ih < p.H && iw < p.W)
                                v = p.src[(size_t)((ci * p.B + b) * p.H + ih) * p.W + iw];
                        }
                    }
                }
                lBp[(n * 2 + (k >> 4)) * 16 + (k & 15)] = (__bf16)v;
            }
            if (full)
                asm volatile("s_wait_asynccnt 0" ::: "memory");
            __syncthreads();
        }

        // ---- fragments (vectorized ds loads) -------------------------------
        const int hi  = lane >> 4;
        const int row = wm * 16 + (lane & 15);
        v8bf a0 = *(const v8bf*)&lA[row * TILE_K + hi * 16];
        v8bf a1 = *(const v8bf*)&lA[row * TILE_K + hi * 16 + 8];
        v16bf afrag = __builtin_shufflevector(a0, a1, 0, 1, 2, 3, 4, 5, 6, 7,
                                              8, 9, 10, 11, 12, 13, 14, 15);
        const int c0 = wn * 32 + (lane & 15);       // tile-local col, subtile 0
        v16bf bfrag0 = *(const v16bf*)&lBp[(c0 * 2 + hi) * 16];
        v16bf bfrag1 = *(const v16bf*)&lBp[((c0 + 16) * 2 + hi) * 16];

        acc0 = __builtin_amdgcn_wmma_f32_16x16x32_bf16(
                   false, afrag, false, bfrag0, (short)0, acc0, false, false);
        acc1 = __builtin_amdgcn_wmma_f32_16x16x32_bf16(
                   false, afrag, false, bfrag1, (short)0, acc1, false, false);
        __syncthreads();
    }

    // ---- epilogue: two 16x16 f32 C tiles (lane 0-15: M=r, 16-31: M=8+r) ----
    const int colA  = n0 + wn * 32 + (lane & 15);
    const int colB  = colA + 16;
    const int rbase = m0 + wm * 16 + ((lane >> 4) ? 8 : 0);
#pragma unroll
    for (int r = 0; r < 8; ++r) {
        int row = rbase + r;
        if (row < p.Co) {
            if (colA < Ntot) p.dst[(size_t)row * Ntot + colA] = acc0[r];
            if (colB < Ntot) p.dst[(size_t)row * Ntot + colB] = acc1[r];
        }
    }
}

// ---- InstanceNorm (affine=False) + PReLU, one block per (c,b) plane --------
__launch_bounds__(256)
__global__ void inorm_prelu_kernel(const float* __restrict__ src,
                                   float* __restrict__ dst,
                                   const float* __restrict__ alpha, int HW) {
    __shared__ float r1[256], r2[256];
    const float* s = src + (size_t)blockIdx.x * HW;
    float*       d = dst + (size_t)blockIdx.x * HW;
    float sum = 0.f, sq = 0.f;
    for (int i = threadIdx.x; i < HW; i += 256) {
        float v = s[i]; sum += v; sq += v * v;
    }
    r1[threadIdx.x] = sum; r2[threadIdx.x] = sq;
    __syncthreads();
    for (int st = 128; st > 0; st >>= 1) {
        if (threadIdx.x < st) {
            r1[threadIdx.x] += r1[threadIdx.x + st];
            r2[threadIdx.x] += r2[threadIdx.x + st];
        }
        __syncthreads();
    }
    float m   = r1[0] / (float)HW;
    float var = r2[0] / (float)HW - m * m;
    float inv = rsqrtf(var + 1e-5f);
    float a   = alpha[0];
    for (int i = threadIdx.x; i < HW; i += 256) {
        float v = (s[i] - m) * inv;
        d[i] = (v >= 0.f) ? v : a * v;
    }
}

// ---- weight converters -----------------------------------------------------
__global__ void cvt_kernel(const float* __restrict__ src, __bf16* __restrict__ dst, int n) {
    int i = blockIdx.x * blockDim.x + threadIdx.x;
    if (i < n) dst[i] = (__bf16)src[i];
}

// convT weights [Ci,Co,3,3] -> A[co, ci*9 + kh*3 + kw] = w[ci,co,2-kh,2-kw]
__global__ void cvtT_kernel(const float* __restrict__ w, __bf16* __restrict__ dst,
                            int Ci, int Co) {
    int i = blockIdx.x * blockDim.x + threadIdx.x;
    int total = Ci * Co * 9;
    if (i >= total) return;
    int co = i / (Ci * 9), r = i % (Ci * 9);
    int ci = r / 9, kk = r % 9;
    int kh = kk / 3, kw = kk % 3;
    dst[i] = (__bf16)w[((size_t)(ci * Co + co) * 3 + (2 - kh)) * 3 + (2 - kw)];
}

__global__ void fill_kernel(float* __restrict__ p, int n, float v) {
    int i = blockIdx.x * blockDim.x + threadIdx.x;
    if (i < n) p[i] = v;
}

// ---- SAGE helpers ----------------------------------------------------------
__global__ void deg_kernel(const int* __restrict__ edges, int nE, float* __restrict__ deg) {
    int e = blockIdx.x * blockDim.x + threadIdx.x;
    if (e < nE) atomicAdd(&deg[edges[nE + e]], 1.0f);
}

// agg[c, b*Nn + tgt] += x[c, b*Nn + src]
__global__ void scatter_kernel(const float* __restrict__ x, const int* __restrict__ edges,
                               int nE, float* __restrict__ agg, int B, int C, int Nn) {
    int t = blockIdx.x * blockDim.x + threadIdx.x;
    if (t >= C * nE) return;
    int c = t / nE, e = t % nE;
    int s = edges[e], g = edges[nE + e];
    for (int b = 0; b < B; ++b)
        atomicAdd(&agg[(size_t)(c * B + b) * Nn + g],
                  x[(size_t)(c * B + b) * Nn + s]);
}

__global__ void aggnorm_kernel(float* __restrict__ agg, const float* __restrict__ deg,
                               int total, int Nn) {
    int i = blockIdx.x * blockDim.x + threadIdx.x;
    if (i >= total) return;
    agg[i] /= fmaxf(deg[i % Nn], 1.0f);
}

__global__ void sage_out_kernel(const float* __restrict__ t1, const float* __restrict__ t2,
                                const float* __restrict__ bl, float* __restrict__ out,
                                int C, int M) {
    int i = blockIdx.x * blockDim.x + threadIdx.x;
    if (i >= C * M) return;
    int c = i / M;
    out[i] = fmaxf(t1[i] + t2[i] + bl[c], 0.0f);
}

// ---- final: [C,B,H,W] + bias -> NCHW d_out --------------------------------
__global__ void final_kernel(const float* __restrict__ raw, const float* __restrict__ bias,
                             float* __restrict__ out, int B, int C, int HW) {
    int i = blockIdx.x * blockDim.x + threadIdx.x;
    if (i >= B * C * HW) return;
    int b = i / (C * HW), r = i % (C * HW);
    int c = r / HW, j = r % HW;
    out[i] = raw[(size_t)(c * B + b) * HW + j] + bias[c];
}

// ---------------------------------------------------------------------------
extern "C" void kernel_launch(void* const* d_in, const int* in_sizes, int n_in,
                              void* d_out, int out_size, void* d_ws, size_t ws_size,
                              hipStream_t stream) {
    const int B = 16;
    const float* x    = (const float*)d_in[0];
    const int*  edges = (const int*)d_in[1];
    const float* w_d4 = (const float*)d_in[2];  const float* a_d4 = (const float*)d_in[4];
    const float* w_d3 = (const float*)d_in[5];  const float* a_d3 = (const float*)d_in[7];
    const float* w_d2 = (const float*)d_in[8];  const float* a_d2 = (const float*)d_in[10];
    const float* w_d1 = (const float*)d_in[11]; const float* a_d1 = (const float*)d_in[13];
    const float* w_bot= (const float*)d_in[14]; const float* a_bot= (const float*)d_in[16];
    const float* wl1  = (const float*)d_in[17]; const float* bl1  = (const float*)d_in[18];
    const float* wr1  = (const float*)d_in[19];
    const float* wl2  = (const float*)d_in[20]; const float* bl2  = (const float*)d_in[21];
    const float* wr2  = (const float*)d_in[22];
    const float* w_u1 = (const float*)d_in[23]; const float* a_u1 = (const float*)d_in[25];
    const float* w_u2 = (const float*)d_in[26]; const float* a_u2 = (const float*)d_in[28];
    const float* w_u3 = (const float*)d_in[29]; const float* a_u3 = (const float*)d_in[31];
    const float* w_u4 = (const float*)d_in[32]; const float* b_u4 = (const float*)d_in[33];

    // ---- workspace carve ---------------------------------------------------
    char* ws = (char*)d_ws;
    size_t off = 0;
    auto alloc_f = [&](size_t n) {
        float* p = (float*)(ws + off); off += ((n * 4 + 255) / 256) * 256; return p;
    };
    auto alloc_b = [&](size_t n) {
        __bf16* p = (__bf16*)(ws + off); off += ((n * 2 + 255) / 256) * 256; return p;
    };

    float* cat3 = alloc_f((size_t)64  * B * 200 * 200); // [act_d4 | u3_out]
    float* cat2 = alloc_f((size_t)128 * B * 100 * 100); // [act_d3 | u2_out]
    float* cat1 = alloc_f((size_t)256 * B * 50  * 50);  // [act_d2 | u1_out]
    float* cat0 = alloc_f((size_t)512 * B * 625);       // [act_d1 | gnn_out]
    float* raw  = alloc_f((size_t)32  * B * 200 * 200); // pre-norm scratch (max 20.48M)
    float* xg   = alloc_f((size_t)256 * B * 625);
    float* xg2  = alloc_f((size_t)256 * B * 625);
    float* agg  = alloc_f((size_t)256 * B * 625);
    float* t1   = alloc_f((size_t)256 * B * 625);
    float* t2   = alloc_f((size_t)256 * B * 625);
    float* deg  = alloc_f(1024);

    __bf16* A_d4  = alloc_b(32  * 9);
    __bf16* A_d3  = alloc_b(64  * 288);
    __bf16* A_d2  = alloc_b(128 * 576);
    __bf16* A_d1  = alloc_b(256 * 1152);
    __bf16* A_bot = alloc_b(256 * 2304);
    __bf16* A_wl1 = alloc_b(256 * 256);
    __bf16* A_wr1 = alloc_b(256 * 256);
    __bf16* A_wl2 = alloc_b(256 * 256);
    __bf16* A_wr2 = alloc_b(256 * 256);
    __bf16* A_u1  = alloc_b(128 * 4608);
    __bf16* A_u2  = alloc_b(64  * 2304);
    __bf16* A_u3  = alloc_b(32  * 1152);
    __bf16* A_u4  = alloc_b(2   * 576);

    auto cvt = [&](const float* s, __bf16* d, int n) {
        cvt_kernel<<<(n + 255) / 256, 256, 0, stream>>>(s, d, n);
    };
    auto cvtT = [&](const float* s, __bf16* d, int Ci, int Co) {
        int n = Ci * Co * 9;
        cvtT_kernel<<<(n + 255) / 256, 256, 0, stream>>>(s, d, Ci, Co);
    };
    auto conv = [&](const float* src, const __bf16* wA, float* dst,
                    int Co, int Ci, int H, int W, int Ho, int Wo,
                    int KH, int stride, int dil, int pad) {
        GemmConvParams p{src, wA, dst, Co, Ci, B, H, W, Ho, Wo, KH, KH, stride, dil, pad};
        int Ntot = B * Ho * Wo;
        dim3 g((Ntot + TILE_N - 1) / TILE_N, (Co + TILE_M - 1) / TILE_M);
        conv_wmma_kernel<<<g, dim3(256), 0, stream>>>(p);
    };
    auto inorm = [&](const float* src, float* dst, const float* alpha, int C, int HW) {
        inorm_prelu_kernel<<<dim3(C * B), dim3(256), 0, stream>>>(src, dst, alpha, HW);
    };

    // ---- weight conversion -------------------------------------------------
    cvt(w_d4,  A_d4,  32 * 9);
    cvt(w_d3,  A_d3,  64 * 288);
    cvt(w_d2,  A_d2,  128 * 576);
    cvt(w_d1,  A_d1,  256 * 1152);
    cvt(w_bot, A_bot, 256 * 2304);
    cvt(wl1, A_wl1, 256 * 256); cvt(wr1, A_wr1, 256 * 256);
    cvt(wl2, A_wl2, 256 * 256); cvt(wr2, A_wr2, 256 * 256);
    cvtT(w_u1, A_u1, 512, 128);
    cvtT(w_u2, A_u2, 256, 64);
    cvtT(w_u3, A_u3, 128, 32);
    cvtT(w_u4, A_u4, 64, 2);

    // ---- encoder (biases cancel inside InstanceNorm) -----------------------
    conv(x,    A_d4, raw, 32, 1,   400, 400, 200, 200, 3, 2, 1, 1);
    inorm(raw, cat3, a_d4, 32, 200 * 200);
    conv(cat3, A_d3, raw, 64, 32,  200, 200, 100, 100, 3, 2, 1, 1);
    inorm(raw, cat2, a_d3, 64, 100 * 100);
    conv(cat2, A_d2, raw, 128, 64, 100, 100, 50, 50, 3, 2, 1, 1);
    inorm(raw, cat1, a_d2, 128, 50 * 50);
    conv(cat1, A_d1, raw, 256, 128, 50, 50, 25, 25, 3, 2, 1, 1);
    inorm(raw, cat0, a_d1, 256, 625);
    conv(cat0, A_bot, raw, 256, 256, 25, 25, 25, 25, 3, 1, 1, 1);
    inorm(raw, xg, a_bot, 256, 625);

    // ---- GraphSAGE ×2 on [256, B*625] --------------------------------------
    const int nE = 5000, Nn = 625, CM = 256 * B * Nn;
    fill_kernel<<<4, 256, 0, stream>>>(deg, 1024, 0.0f);
    deg_kernel<<<(nE + 255) / 256, 256, 0, stream>>>(edges, nE, deg);

    fill_kernel<<<(CM + 255) / 256, 256, 0, stream>>>(agg, CM, 0.0f);
    scatter_kernel<<<(256 * nE + 255) / 256, 256, 0, stream>>>(xg, edges, nE, agg, B, 256, Nn);
    aggnorm_kernel<<<(CM + 255) / 256, 256, 0, stream>>>(agg, deg, CM, Nn);
    conv(agg, A_wl1, t1, 256, 256, 25, 25, 25, 25, 1, 1, 1, 0);
    conv(xg,  A_wr1, t2, 256, 256, 25, 25, 25, 25, 1, 1, 1, 0);
    sage_out_kernel<<<(CM + 255) / 256, 256, 0, stream>>>(t1, t2, bl1, xg2, 256, B * Nn);

    fill_kernel<<<(CM + 255) / 256, 256, 0, stream>>>(agg, CM, 0.0f);
    scatter_kernel<<<(256 * nE + 255) / 256, 256, 0, stream>>>(xg2, edges, nE, agg, B, 256, Nn);
    aggnorm_kernel<<<(CM + 255) / 256, 256, 0, stream>>>(agg, deg, CM, Nn);
    conv(agg, A_wl2, t1, 256, 256, 25, 25, 25, 25, 1, 1, 1, 0);
    conv(xg2, A_wr2, t2, 256, 256, 25, 25, 25, 25, 1, 1, 1, 0);
    sage_out_kernel<<<(CM + 255) / 256, 256, 0, stream>>>(t1, t2, bl2,
                                                          cat0 + (size_t)256 * B * 625,
                                                          256, B * Nn);

    // ---- decoder (ConvTranspose == lhs-dilated conv of flipped weights) ----
    conv(cat0, A_u1, raw, 128, 512, 25, 25, 50, 50, 3, 1, 2, 1);
    inorm(raw, cat1 + (size_t)128 * B * 2500, a_u1, 128, 2500);
    conv(cat1, A_u2, raw, 64, 256, 50, 50, 100, 100, 3, 1, 2, 1);
    inorm(raw, cat2 + (size_t)64 * B * 10000, a_u2, 64, 10000);
    conv(cat2, A_u3, raw, 32, 128, 100, 100, 200, 200, 3, 1, 2, 1);
    inorm(raw, cat3 + (size_t)32 * B * 40000, a_u3, 32, 40000);
    conv(cat3, A_u4, raw, 2, 64, 200, 200, 400, 400, 3, 1, 2, 1);

    int tot = B * 2 * 400 * 400;
    final_kernel<<<(tot + 255) / 256, 256, 0, stream>>>(raw, b_u4, (float*)d_out,
                                                        B, 2, 400 * 400);
    (void)in_sizes; (void)n_in; (void)out_size; (void)ws_size;
}
